// QuanvolutionGraphHybrid_65481071399731
// MI455X (gfx1250) — matrix operand order repaired and checked by reading, not compiled, not run on hardware
//
#include <hip/hip_runtime.h>
#include <hip/hip_bf16.h>
#include <math.h>

// ---------------------------------------------------------------------------
// QuanvolutionGraphHybrid for MI455X (gfx1250, wave32, WMMA, async->LDS)
//   out[0:81920]              = log_probs  [8192,10] f32
//   out[81920: +8192*8192]    = adj        [8192,8192] f32
// Workspace: qf f32 [8192,784] (25.7MB) then qn bf16 [8192,800] (13.1MB)
// ---------------------------------------------------------------------------

typedef __attribute__((ext_vector_type(16))) __bf16 v16bf;
typedef __attribute__((ext_vector_type(8)))  float  v8f;

#define NROWS   8192
#define DQF     784
#define KPAD    800          // 784 padded to multiple of 32 (zero-filled)
#define NT      (KPAD / 32)  // 25 K-steps
#define THRESH  0.9f
#define LDSS    40           // LDS row stride in ushorts (32 + 8 pad), 80B

__device__ __forceinline__ unsigned short f2bf(float f) {
    unsigned int u = __float_as_uint(f);
    u += 0x7FFFu + ((u >> 16) & 1u);   // round-to-nearest-even
    return (unsigned short)(u >> 16);
}

// Direct global -> LDS async copy (CDNA5, ASYNCcnt-tracked, GV mode).
// lds_off: wave-relative LDS byte offset = low 32 bits of the flat pointer
// to a __shared__ object (ISA 10.2: LDS aperture keeps offset in addr[31:0]).
__device__ __forceinline__ void async_ld_b128(void* lds_ptr, const void* gptr) {
    unsigned int lds_off = (unsigned int)(unsigned long long)lds_ptr;
    asm volatile("global_load_async_to_lds_b128 %0, %1, off"
                 :: "v"(lds_off), "v"(gptr) : "memory");
}

// ---------------------------------------------------------------------------
// Kernel 1: fused 2x2/s2 conv (+bias) and 4x4 unitary per patch-group.
// flat[b, c*196 + p]; group n -> c = n/49, positions p0=4*(n%49)..p0+3.
// qf[b, 4n+w] = sum_k conv[b,c,p0+k] * U[w,k]
// ---------------------------------------------------------------------------
__global__ void k_conv_unitary(const float* __restrict__ x,
                               const float* __restrict__ cw,
                               const float* __restrict__ cb,
                               const float* __restrict__ U,
                               float* __restrict__ qf) {
    int t = blockIdx.x * blockDim.x + threadIdx.x;
    int b = t / 196;
    int n = t - b * 196;
    if (b >= NROWS) return;
    int c = n / 49;
    int q = n - c * 49;
    float w00 = cw[c*4+0], w01 = cw[c*4+1], w10 = cw[c*4+2], w11 = cw[c*4+3];
    float bias = cb[c];
    float v[4];
#pragma unroll
    for (int k = 0; k < 4; ++k) {
        int p = 4*q + k;
        int i = p / 14, j = p - i*14;
        const float* px = x + (size_t)b*784 + (2*i)*28 + 2*j;
        v[k] = w00*px[0] + w01*px[1] + w10*px[28] + w11*px[29] + bias;
    }
    float* o = qf + (size_t)b*DQF + n*4;
#pragma unroll
    for (int w = 0; w < 4; ++w) {
        o[w] = v[0]*U[w*4+0] + v[1]*U[w*4+1] + v[2]*U[w*4+2] + v[3]*U[w*4+3];
    }
}

// ---------------------------------------------------------------------------
// Kernel 2: per-row head. One block (256 thr) per row:
//   norm^2 + 10 logits via LDS reduction; log_softmax -> d_out;
//   qn_bf16[b,k] = bf16(qf[b,k]/(||qf||+eps)), zero-padded to K=800.
// ---------------------------------------------------------------------------
__global__ __launch_bounds__(256)
void k_head(const float* __restrict__ qf,
            const float* __restrict__ lw,
            const float* __restrict__ lb,
            float* __restrict__ out_logp,
            unsigned short* __restrict__ qn) {
    int b = blockIdx.x;
    int tid = threadIdx.x;
    const float* row = qf + (size_t)b * DQF;

    float acc[11];
#pragma unroll
    for (int j = 0; j < 11; ++j) acc[j] = 0.f;
    for (int k = tid; k < DQF; k += 256) {
        float qv = row[k];
        acc[10] += qv * qv;
#pragma unroll
        for (int j = 0; j < 10; ++j) acc[j] += qv * lw[(size_t)j*DQF + k];
    }

    __shared__ float red[11 * 256];
#pragma unroll
    for (int j = 0; j < 11; ++j) red[j*256 + tid] = acc[j];
    __syncthreads();
    for (int s = 128; s > 0; s >>= 1) {
        if (tid < s) {
#pragma unroll
            for (int j = 0; j < 11; ++j) red[j*256 + tid] += red[j*256 + tid + s];
        }
        __syncthreads();
    }

    __shared__ float s_rnorm;
    if (tid == 0) {
        float l[10];
        float mx = -1e30f;
#pragma unroll
        for (int j = 0; j < 10; ++j) { l[j] = red[j*256] + lb[j]; mx = fmaxf(mx, l[j]); }
        float se = 0.f;
#pragma unroll
        for (int j = 0; j < 10; ++j) se += expf(l[j] - mx);
        float lse = logf(se);
#pragma unroll
        for (int j = 0; j < 10; ++j) out_logp[(size_t)b*10 + j] = l[j] - mx - lse;
        s_rnorm = 1.0f / (sqrtf(red[10*256]) + 1e-12f);
    }
    __syncthreads();
    float rn = s_rnorm;
    unsigned short* qrow = qn + (size_t)b * KPAD;
    for (int k = tid; k < KPAD; k += 256) {
        qrow[k] = (k < DQF) ? f2bf(row[k] * rn) : (unsigned short)0;
    }
}

// ---------------------------------------------------------------------------
// Kernel 3: adj = ((qn @ qn^T)^2 >= 0.9) with zero diagonal.
// Block tile 128x128, 8 waves (wave32); each wave: 2x4 tiles of 16x16.
// K stepped by 32 (bf16 WMMA, f32 accum). Double-buffered LDS fed by
// GLOBAL_LOAD_ASYNC_TO_LDS_B128 (ASYNCcnt), overlapping copy with WMMA.
// Fragment gathers follow CDNA5 ISA 7.12.2 VGPR layouts.
// ---------------------------------------------------------------------------
__global__ __launch_bounds__(256)
void k_gram_adj(const unsigned short* __restrict__ qn,
                float* __restrict__ adj) {
    __shared__ __align__(16) unsigned short lds_a[2][128][LDSS];
    __shared__ __align__(16) unsigned short lds_b[2][128][LDSS];

    const int mbase = blockIdx.y * 128;
    const int nbase = blockIdx.x * 128;
    const int tid   = threadIdx.x;
    const int lane  = tid & 31;
    const int wave  = tid >> 5;
    const int m0    = (wave & 3) * 32;   // wave's m offset in block tile
    const int n0    = (wave >> 2) * 64;  // wave's n offset in block tile
    const int g     = lane >> 4;         // lane half (ISA layout group)
    const int lm    = lane & 15;

    // tile staging: thread -> (row r, 32B half h) of the 128x64B tile
    const int r = tid >> 1;
    const int h = tid & 1;
    const unsigned short* ga = qn + (size_t)(mbase + r) * KPAD + h * 16;
    const unsigned short* gb = qn + (size_t)(nbase + r) * KPAD + h * 16;

    const v8f vzero = {0.f, 0.f, 0.f, 0.f, 0.f, 0.f, 0.f, 0.f};
    v8f acc[2][4];
#pragma unroll
    for (int mi = 0; mi < 2; ++mi)
#pragma unroll
        for (int ni = 0; ni < 4; ++ni) acc[mi][ni] = vzero;

    // Prologue: stage tile 0 into buffer 0 (4 async b128 copies / thread).
    async_ld_b128(&lds_a[0][r][h*16    ], ga);
    async_ld_b128(&lds_a[0][r][h*16 + 8], ga + 8);
    async_ld_b128(&lds_b[0][r][h*16    ], gb);
    async_ld_b128(&lds_b[0][r][h*16 + 8], gb + 8);

    for (int kk = 0; kk < NT; ++kk) {
        const int cur = kk & 1;
        if (kk + 1 < NT) {
            // Prefetch next K-tile into the other buffer (stays in flight).
            const int k1 = (kk + 1) * 32;
            const int nxt = cur ^ 1;
            async_ld_b128(&lds_a[nxt][r][h*16    ], ga + k1);
            async_ld_b128(&lds_a[nxt][r][h*16 + 8], ga + k1 + 8);
            async_ld_b128(&lds_b[nxt][r][h*16    ], gb + k1);
            async_ld_b128(&lds_b[nxt][r][h*16 + 8], gb + k1 + 8);
            // our 4 current-tile copies done; next tile's 4 still in flight
            asm volatile("s_wait_asynccnt 0x4" ::: "memory");
        } else {
            asm volatile("s_wait_asynccnt 0x0" ::: "memory");
        }
        __syncthreads();   // current buffer fully written by all waves

        union { uint4 q[2]; v16bf v; } fa[2], fb[4];
        // A (16x32 bf16): lane lm = M row; chunks K=[8g,8g+8) and [16+8g,+8)
#pragma unroll
        for (int mi = 0; mi < 2; ++mi) {
            const unsigned short* p = &lds_a[cur][m0 + mi*16 + lm][0];
            fa[mi].q[0] = *(const uint4*)(p + 8*g);
            fa[mi].q[1] = *(const uint4*)(p + 16 + 8*g);
        }
    // B (32x16 bf16): lane lm = N col; chunks K=[8g,8g+8) and [16+8g,+8)
#pragma unroll
        for (int ni = 0; ni < 4; ++ni) {
            const unsigned short* p = &lds_b[cur][n0 + ni*16 + lm][0];
            fb[ni].q[0] = *(const uint4*)(p + 8*g);
            fb[ni].q[1] = *(const uint4*)(p + 16 + 8*g);
        }
#pragma unroll
        for (int mi = 0; mi < 2; ++mi)
#pragma unroll
            for (int ni = 0; ni < 4; ++ni)
                acc[mi][ni] = __builtin_amdgcn_wmma_f32_16x16x32_bf16(
                    false, fa[mi].v, false, fb[ni].v,
                    (short)0, acc[mi][ni], false, false);

        __syncthreads();   // reads of buf[cur] done before it is re-issued
    }

    // Epilogue: C/D layout — VGPR rr: m = rr + 8*g, n = lane%16
#pragma unroll
    for (int mi = 0; mi < 2; ++mi) {
#pragma unroll
        for (int ni = 0; ni < 4; ++ni) {
#pragma unroll
            for (int rr = 0; rr < 8; ++rr) {
                int gm = mbase + m0 + mi*16 + rr + 8*g;
                int gn = nbase + n0 + ni*16 + lm;
                float c = acc[mi][ni][rr];
                float fid = c * c;
                adj[(size_t)gm * NROWS + gn] =
                    (fid >= THRESH && gm != gn) ? 1.0f : 0.0f;
            }
        }
    }
}

// ---------------------------------------------------------------------------
extern "C" void kernel_launch(void* const* d_in, const int* in_sizes, int n_in,
                              void* d_out, int out_size, void* d_ws, size_t ws_size,
                              hipStream_t stream) {
    (void)in_sizes; (void)n_in; (void)out_size; (void)ws_size;
    const float* x  = (const float*)d_in[0];   // [8192,1,28,28]
    const float* cw = (const float*)d_in[1];   // [4,1,2,2]
    const float* cb = (const float*)d_in[2];   // [4]
    const float* U  = (const float*)d_in[3];   // [4,4]
    const float* lw = (const float*)d_in[4];   // [10,784]
    const float* lb = (const float*)d_in[5];   // [10]

    float* out_logp = (float*)d_out;                       // [8192,10]
    float* out_adj  = out_logp + (size_t)NROWS * 10;       // [8192,8192]

    char* ws = (char*)d_ws;
    float* qf = (float*)ws;                                          // 25.7 MB
    unsigned short* qn =
        (unsigned short*)(ws + (size_t)NROWS * DQF * sizeof(float)); // 13.1 MB

    k_conv_unitary<<<(NROWS * 196) / 256, 256, 0, stream>>>(x, cw, cb, U, qf);
    k_head<<<NROWS, 256, 0, stream>>>(qf, lw, lb, out_logp, qn);
    dim3 grid(NROWS / 128, NROWS / 128);
    k_gram_adj<<<grid, 256, 0, stream>>>(qn, out_adj);
}
